// DGCNN_2121713844358
// MI455X (gfx1250) — matrix operand order repaired and checked by reading, not compiled, not run on hardware
//
#include <hip/hip_runtime.h>

// Problem sizes (fixed by reference)
#define NB    4096
#define NN    64
#define CIN   128
#define KH    4
#define COUT  64
#define NCLS  3

typedef __bf16 bf16_t;
typedef __attribute__((ext_vector_type(4)))  __bf16 v4bf;
typedef __attribute__((ext_vector_type(8)))  __bf16 v8bf;
typedef __attribute__((ext_vector_type(16))) __bf16 v16bf;
typedef __attribute__((ext_vector_type(8)))  float  v8f;
typedef __attribute__((ext_vector_type(4)))  float  v4f;

// Load a 16-element bf16 WMMA operand from two 16B-aligned LDS chunks.
__device__ __forceinline__ v16bf ld_pair(const bf16_t* p0, const bf16_t* p1) {
  v8bf lo = *(const v8bf*)p0;
  v8bf hi = *(const v8bf*)p1;
  return __builtin_shufflevector(lo, hi, 0,1,2,3,4,5,6,7,8,9,10,11,12,13,14,15);
}

__device__ __forceinline__ v8f wmma_bf16(v16bf a, v16bf b, v8f c) {
  // (neg_a, A, neg_b, B, c_mod, C, reuse_a, reuse_b)
  return __builtin_amdgcn_wmma_f32_16x16x32_bf16(false, a, false, b, (short)0, c, false, false);
}

// ---------------------------------------------------------------------------
// Prep kernel (1 block): build T_cat bf16 [64][256] (cols j = k*64+m) and
// W^T bf16 [256][128] (row = k*64+o, col = c) in the workspace.
// ---------------------------------------------------------------------------
__global__ __launch_bounds__(256) void dgcnn_prep(const float* __restrict__ A,
                                                  const float* __restrict__ gcw,
                                                  bf16_t* __restrict__ Tg,
                                                  bf16_t* __restrict__ Wtg) {
  __shared__ float sLn[NN][NN];   // holds Ar, then Ln (in place)
  __shared__ float sT2[NN][NN];
  __shared__ float sT3[NN][NN];
  __shared__ float sd[NN];
  const int tid = threadIdx.x;

  // Ar = relu(A) with zero diagonal, symmetrized
  for (int idx = tid; idx < NN * NN; idx += 256) {
    int n = idx >> 6, m = idx & 63;
    float v = (n == m) ? 0.0f
                       : fmaxf(A[n * NN + m], 0.0f) + fmaxf(A[m * NN + n], 0.0f);
    sLn[n][m] = v;
  }
  __syncthreads();
  if (tid < NN) {
    float s = 0.0f;
    for (int m = 0; m < NN; ++m) s += sLn[tid][m];
    sd[tid] = rsqrtf(s + 1e-10f);
  }
  __syncthreads();
  // Ln = L - I = -(d_n * Ar * d_m)
  for (int idx = tid; idx < NN * NN; idx += 256) {
    int n = idx >> 6, m = idx & 63;
    sLn[n][m] = -sd[n] * sLn[n][m] * sd[m];
  }
  __syncthreads();
  // T2 = 2*Ln@Ln - I
  for (int idx = tid; idx < NN * NN; idx += 256) {
    int n = idx >> 6, m = idx & 63;
    float s = 0.0f;
    for (int j = 0; j < NN; ++j) s = fmaf(sLn[n][j], sLn[j][m], s);
    sT2[n][m] = 2.0f * s - ((n == m) ? 1.0f : 0.0f);
  }
  __syncthreads();
  // T3 = 2*Ln@T2 - Ln
  for (int idx = tid; idx < NN * NN; idx += 256) {
    int n = idx >> 6, m = idx & 63;
    float s = 0.0f;
    for (int j = 0; j < NN; ++j) s = fmaf(sLn[n][j], sT2[j][m], s);
    sT3[n][m] = 2.0f * s - sLn[n][m];
  }
  __syncthreads();
  // T_cat[n][k*64+m]
  for (int idx = tid; idx < NN * NN; idx += 256) {
    int n = idx >> 6, m = idx & 63;
    Tg[n * 256 +   0 + m] = (bf16_t)((n == m) ? 1.0f : 0.0f);
    Tg[n * 256 +  64 + m] = (bf16_t)sLn[n][m];
    Tg[n * 256 + 128 + m] = (bf16_t)sT2[n][m];
    Tg[n * 256 + 192 + m] = (bf16_t)sT3[n][m];
  }
  // W^T[(k*64+o)][c] = gc_w[k][c][o]
  for (int idx = tid; idx < 256 * CIN; idx += 256) {
    int row = idx >> 7, c = idx & 127;
    int k = row >> 6, o = row & 63;
    Wtg[idx] = (bf16_t)gcw[((k << 7) | c) * 64 + o];
  }
}

// ---------------------------------------------------------------------------
// Main fused kernel: one block per batch. 256 threads = 8 waves.
// LDS (dynamic, aliased):
//   region A [0, 33792): xs bf16[64][136] -> Tb bf16[64][264] -> hs f32[64][68]
//   region B [33792, 103424): wsT bf16[256][136] -> Yt bf16[64][264]
//   red f32[768] at [103424, 106496)
// ---------------------------------------------------------------------------
#define OFF_B   33792
#define OFF_RED 103424
#define SMEM_BYTES 106496

__global__ __launch_bounds__(256) void dgcnn_main(const float* __restrict__ x,
                                                  const bf16_t* __restrict__ Tg,
                                                  const bf16_t* __restrict__ Wtg,
                                                  const float* __restrict__ fcw,
                                                  const float* __restrict__ fcb,
                                                  float* __restrict__ out) {
  extern __shared__ __align__(32) unsigned char smem[];
  bf16_t (*xs)[136]  = (bf16_t(*)[136])(smem);
  bf16_t (*Tb)[264]  = (bf16_t(*)[264])(smem);
  float  (*hs)[68]   = (float (*)[68])(smem);
  bf16_t (*wsT)[136] = (bf16_t(*)[136])(smem + OFF_B);
  bf16_t (*Yt)[264]  = (bf16_t(*)[264])(smem + OFF_B);
  float* red         = (float*)(smem + OFF_RED);

  const int tid  = threadIdx.x;
  const int wv   = tid >> 5;
  const int lane = tid & 31;
  const int ln   = lane & 15;
  const int half = lane >> 4;
  const int b    = blockIdx.x;

  // ---- stage x[b] (HBM, read once; float4 loads -> bf16x4 LDS stores) ----
  {
    const v4f* xb4 = (const v4f*)(x + (size_t)b * (NN * CIN));
    // 64*128/4 = 2048 float4 chunks; 8 per thread; 32 chunks per row
    for (int idx = tid; idx < (NN * CIN) / 4; idx += 256) {
      v4f f = xb4[idx];
      v4bf h;
      h[0] = (bf16_t)f[0]; h[1] = (bf16_t)f[1];
      h[2] = (bf16_t)f[2]; h[3] = (bf16_t)f[3];
      *(v4bf*)&xs[idx >> 5][(idx & 31) * 4] = h;
    }
    // ---- stage W^T (L2-resident bf16; 16B chunks) ----
    const v8bf* wg8 = (const v8bf*)Wtg;
    // 256*128/8 = 4096 chunks; 16 per thread; 16 chunks per row
    for (int idx = tid; idx < (256 * CIN) / 8; idx += 256)
      *(v8bf*)&wsT[idx >> 4][(idx & 15) * 8] = wg8[idx];
  }
  __syncthreads();

  // ---- GEMM1: Y[64x256] = x_b[64x128] @ W_cat[128x256] ----
  // wave wv: M-tile mt = wv>>1 ; N-tiles ntb..ntb+7
  const v8f vzero = {0.f, 0.f, 0.f, 0.f, 0.f, 0.f, 0.f, 0.f};
  v8f acc[8];
#pragma unroll
  for (int t = 0; t < 8; ++t) acc[t] = vzero;
  const int mt  = wv >> 1;
  const int ntb = (wv & 1) * 8;
#pragma unroll
  for (int kc = 0; kc < 4; ++kc) {
    const bf16_t* ar = &xs[mt * 16 + ln][kc * 32 + half * 16];
    v16bf a = ld_pair(ar, ar + 8);
    // software-pipeline the B operand so DS loads overlap WMMA
    const bf16_t* br0 = &wsT[ntb * 16 + ln][kc * 32 + half * 16];
    v16bf bcur = ld_pair(br0, br0 + 8);
#pragma unroll
    for (int t = 0; t < 8; ++t) {
      v16bf bnext = bcur;
      if (t < 7) {
        const bf16_t* br = &wsT[(ntb + t + 1) * 16 + ln][kc * 32 + half * 16];
        bnext = ld_pair(br, br + 8);
      }
      acc[t] = wmma_bf16(a, bcur, acc[t]);
      bcur = bnext;
    }
  }
  __syncthreads();  // done reading xs (region A) and wsT (region B)

  // ---- scatter Y^T (bf16) into region B: Yt[o][k*64+m] = Y[m][k*64+o] ----
#pragma unroll
  for (int t = 0; t < 8; ++t) {
    int q  = (ntb + t) * 16 + ln;  // column of Y = k*64+o
    int o  = q & 63;
    int kb = q - o;                // k*64
#pragma unroll
    for (int v = 0; v < 8; ++v) {
      int m = mt * 16 + half * 8 + v;
      Yt[o][kb + m] = (bf16_t)acc[t][v];
    }
  }
  // stage T_cat into region A (16B chunks; 32 chunks per 256-wide row)
  {
    const v8bf* tg8 = (const v8bf*)Tg;
    for (int idx = tid; idx < (NN * 256) / 8; idx += 256)
      *(v8bf*)&Tb[idx >> 5][(idx & 31) * 8] = tg8[idx];
  }
  __syncthreads();

  // ---- GEMM2: h[64x64] = T_cat[64x256] @ Y_stack[256x64] ----
  v8f acc2[2];
  acc2[0] = vzero;
  acc2[1] = vzero;
  const int mt2 = wv >> 1;
  const int otb = (wv & 1) * 2;
#pragma unroll
  for (int kc = 0; kc < 8; ++kc) {
    const bf16_t* ar = &Tb[mt2 * 16 + ln][kc * 32 + half * 16];
    v16bf a = ld_pair(ar, ar + 8);
    const bf16_t* b0 = &Yt[otb * 16 + ln][kc * 32 + half * 16];
    const bf16_t* b1 = &Yt[(otb + 1) * 16 + ln][kc * 32 + half * 16];
    v16bf bm0 = ld_pair(b0, b0 + 8);
    v16bf bm1 = ld_pair(b1, b1 + 8);
    acc2[0] = wmma_bf16(a, bm0, acc2[0]);
    acc2[1] = wmma_bf16(a, bm1, acc2[1]);
  }
  __syncthreads();  // done reading Tb (region A) before hs overwrite

  // ---- ReLU, store h to LDS (f32) ----
#pragma unroll
  for (int t = 0; t < 2; ++t) {
    int o = (otb + t) * 16 + ln;
#pragma unroll
    for (int v = 0; v < 8; ++v) {
      int n = mt2 * 16 + half * 8 + v;
      hs[n][o] = fmaxf(acc2[t][v], 0.0f);
    }
  }
  __syncthreads();

  // ---- classifier: out[b][c] = sum_i h_flat[i]*fc_w[i][c] + fc_b[c] ----
  float p0 = 0.f, p1 = 0.f, p2 = 0.f;
  const int ibase = tid * 16;
#pragma unroll
  for (int ii = 0; ii < 16; ++ii) {
    int i = ibase + ii;
    float hv = hs[i >> 6][i & 63];
    p0 = fmaf(hv, fcw[i * 3 + 0], p0);
    p1 = fmaf(hv, fcw[i * 3 + 1], p1);
    p2 = fmaf(hv, fcw[i * 3 + 2], p2);
  }
  red[tid] = p0;
  red[256 + tid] = p1;
  red[512 + tid] = p2;
  __syncthreads();
  for (int s = 128; s > 0; s >>= 1) {
    if (tid < s) {
      red[tid]       += red[tid + s];
      red[256 + tid] += red[256 + tid + s];
      red[512 + tid] += red[512 + tid + s];
    }
    __syncthreads();
  }
  if (tid < NCLS) out[b * NCLS + tid] = red[tid * 256] + fcb[tid];
}

// ---------------------------------------------------------------------------
extern "C" void kernel_launch(void* const* d_in, const int* in_sizes, int n_in,
                              void* d_out, int out_size, void* d_ws, size_t ws_size,
                              hipStream_t stream) {
  const float* x   = (const float*)d_in[0];   // [4096,64,128]
  const float* A   = (const float*)d_in[1];   // [64,64]
  const float* gcw = (const float*)d_in[2];   // [4,128,64]
  const float* fcw = (const float*)d_in[3];   // [4096,3]
  const float* fcb = (const float*)d_in[4];   // [3]
  float* out = (float*)d_out;                 // [4096,3]

  bf16_t* Tg  = (bf16_t*)d_ws;                          // 64*256*2  = 32 KB
  bf16_t* Wtg = (bf16_t*)((char*)d_ws + NN * 256 * 2);  // 256*128*2 = 64 KB

  (void)in_sizes; (void)n_in; (void)out_size; (void)ws_size;
  hipFuncSetAttribute((const void*)dgcnn_main,
                      hipFuncAttributeMaxDynamicSharedMemorySize, SMEM_BYTES);

  dgcnn_prep<<<1, 256, 0, stream>>>(A, gcw, Tg, Wtg);
  dgcnn_main<<<NB, 256, SMEM_BYTES, stream>>>(x, Tg, Wtg, fcw, fcb, out);
}
